// MoELayer_71528385347981
// MI455X (gfx1250) — compile-verified, matrix operand order checked by reading, not run on hardware
//
#include <hip/hip_runtime.h>
#include <hip/hip_bf16.h>

typedef __bf16 bf16_t;
typedef __attribute__((ext_vector_type(16))) __bf16 v16bf;
typedef __attribute__((ext_vector_type(8)))  __bf16 v8bf;
typedef __attribute__((ext_vector_type(8)))  float  v8f;

#define T_TOK   4096
#define HID     1024
#define INTER_  4096
#define NEXP    8

// GEMM tiling: 256 threads = 8 waves (4M x 2N), wave tile 64x64 (4x4 wmma tiles)
#define BM  256
#define BN  128
#define BK  64
#define LDT 72   // LDS row stride in bf16 (144B = 9*16B: aligned b128, conflict-free frags)

// ---------------------------------------------------------------------------
// Branchless fast GELU (tanh approximation, no libm divergence)
// ---------------------------------------------------------------------------
__device__ __forceinline__ float gelu_f(float v) {
    float u = 0.7978845608028654f * v * (1.f + 0.044715f * v * v);
    float a = __builtin_fabsf(u);
    float e = __expf(-2.f * a);                       // v_exp_f32 path
    float t = (1.f - e) * __builtin_amdgcn_rcpf(1.f + e);
    t = __builtin_copysignf(t, u);
    return 0.5f * v * (1.f + t);
}

__device__ __forceinline__ v16bf frag16(const bf16_t* p) {
    v8bf lo = *(const v8bf*)p;
    v8bf hi = *(const v8bf*)(p + 16);
    return __builtin_shufflevector(lo, hi, 0,1,2,3,4,5,6,7,8,9,10,11,12,13,14,15);
}

// ---------------------------------------------------------------------------
// Gating: logits = x @ Wg, softmax(8), top-2 (low-index tie break),
// renormalized gates -> combine; raw probs -> ws; X converted to bf16.
// One block (256 thr) per token. No atomics -> deterministic.
// ---------------------------------------------------------------------------
__global__ __launch_bounds__(256) void moe_gating(
    const float* __restrict__ x, const float* __restrict__ Wg,
    bf16_t* __restrict__ Xbf, float* __restrict__ combine,
    float* __restrict__ probs_out)
{
    __shared__ float wsum[8][NEXP];
    __shared__ float logits[NEXP];
    __shared__ float prob[NEXP];

    const int t   = blockIdx.x;
    const int tid = threadIdx.x;
    float p[NEXP];
#pragma unroll
    for (int e = 0; e < NEXP; ++e) p[e] = 0.f;

    for (int h = tid; h < HID; h += 256) {
        float xv = x[t * HID + h];
        Xbf[t * HID + h] = (bf16_t)xv;
        const float4* wr = (const float4*)(Wg + h * NEXP);
        float4 w0 = wr[0], w1 = wr[1];
        p[0] += xv * w0.x; p[1] += xv * w0.y; p[2] += xv * w0.z; p[3] += xv * w0.w;
        p[4] += xv * w1.x; p[5] += xv * w1.y; p[6] += xv * w1.z; p[7] += xv * w1.w;
    }
#pragma unroll
    for (int off = 16; off > 0; off >>= 1)
#pragma unroll
        for (int e = 0; e < NEXP; ++e) p[e] += __shfl_xor(p[e], off, 32);

    const int wv = tid >> 5, ln = tid & 31;
    if (ln == 0)
#pragma unroll
        for (int e = 0; e < NEXP; ++e) wsum[wv][e] = p[e];
    __syncthreads();
    if (tid < NEXP) {
        float s = 0.f;
#pragma unroll
        for (int w = 0; w < 8; ++w) s += wsum[w][tid];
        logits[tid] = s;
    }
    __syncthreads();
    if (tid == 0) {
        float mx = logits[0];
#pragma unroll
        for (int e = 1; e < NEXP; ++e) mx = fmaxf(mx, logits[e]);
        float den = 0.f;
#pragma unroll
        for (int e = 0; e < NEXP; ++e) { prob[e] = __expf(logits[e] - mx); den += prob[e]; }
        float inv = 1.f / den;
#pragma unroll
        for (int e = 0; e < NEXP; ++e) prob[e] *= inv;
        int i0 = 0;
#pragma unroll
        for (int e = 1; e < NEXP; ++e) if (prob[e] > prob[i0]) i0 = e;
        int i1 = (i0 == 0) ? 1 : 0;
#pragma unroll
        for (int e = 0; e < NEXP; ++e)
            if (e != i0 && prob[e] > prob[i1]) i1 = e;
        float g0 = prob[i0], g1 = prob[i1];
        float gs = 1.f / (g0 + g1);
#pragma unroll
        for (int e = 0; e < NEXP; ++e) {
            combine[t * NEXP + e]   = 0.f;
            probs_out[t * NEXP + e] = prob[e];
        }
        combine[t * NEXP + i0] = g0 * gs;
        combine[t * NEXP + i1] = g1 * gs;
    }
}

// ---------------------------------------------------------------------------
// Aux loss: E * sum_e( frac_e * mean_p_e ) * LB_WEIGHT, fixed-order reduction.
// ---------------------------------------------------------------------------
__global__ __launch_bounds__(256) void moe_aux(
    const float* __restrict__ probs, const float* __restrict__ combine,
    float* __restrict__ aux_out)
{
    __shared__ float sp[8][NEXP];
    __shared__ float sc[8][NEXP];
    const int tid = threadIdx.x;
    float lp[NEXP], lc[NEXP];
#pragma unroll
    for (int e = 0; e < NEXP; ++e) { lp[e] = 0.f; lc[e] = 0.f; }
    for (int t = tid; t < T_TOK; t += 256) {
#pragma unroll
        for (int e = 0; e < NEXP; ++e) {
            lp[e] += probs[t * NEXP + e];
            lc[e] += (combine[t * NEXP + e] > 0.f) ? 1.f : 0.f;
        }
    }
#pragma unroll
    for (int off = 16; off > 0; off >>= 1)
#pragma unroll
        for (int e = 0; e < NEXP; ++e) {
            lp[e] += __shfl_xor(lp[e], off, 32);
            lc[e] += __shfl_xor(lc[e], off, 32);
        }
    const int wv = tid >> 5, ln = tid & 31;
    if (ln == 0)
#pragma unroll
        for (int e = 0; e < NEXP; ++e) { sp[wv][e] = lp[e]; sc[wv][e] = lc[e]; }
    __syncthreads();
    if (tid == 0) {
        float aux = 0.f;
#pragma unroll
        for (int e = 0; e < NEXP; ++e) {
            float ps = 0.f, cs = 0.f;
#pragma unroll
            for (int w = 0; w < 8; ++w) { ps += sp[w][e]; cs += sc[w][e]; }
            aux += (cs / (float)(T_TOK * 2)) * (ps / (float)T_TOK);
        }
        aux_out[0] = (float)NEXP * aux * 0.01f;
    }
}

// ===========================================================================
// GEMM core pieces (shared by both GEMM kernels via inlining)
//   A: bf16 row-major [M x K] (lda), staged 256x64 halfs
//   B: fp32 row-major [K x N] (ldb), converted->bf16, LDS-transposed [N][K]
//   Double-buffered LDS, register prefetch, BK=64 (2 wmma K-steps / stage)
// ===========================================================================
struct StageRegs {
    uint4  a[8];
    float4 b[8];
};

__device__ __forceinline__ void load_stage(
    StageRegs& s, const bf16_t* __restrict__ A, int lda,
    const float* __restrict__ W, int ldb,
    int row0, int col0, int k0, int tid)
{
#pragma unroll
    for (int i = 0; i < 8; ++i) {            // A: 256 rows x 64 halfs = 2048 x 16B
        int c = tid + i * 256;
        int r = c >> 3, o = (c & 7) * 8;
        s.a[i] = *(const uint4*)(&A[(size_t)(row0 + r) * lda + k0 + o]);
    }
#pragma unroll
    for (int i = 0; i < 8; ++i) {            // B: 64 k-rows x 128 n fp32
        int c = tid + i * 256;
        int k = c >> 5, n = (c & 31) * 4;
        s.b[i] = *(const float4*)(&W[(size_t)(k0 + k) * ldb + col0 + n]);
    }
}

__device__ __forceinline__ void store_stage(
    const StageRegs& s, bf16_t* __restrict__ As, bf16_t* __restrict__ Bs, int tid)
{
#pragma unroll
    for (int i = 0; i < 8; ++i) {
        int c = tid + i * 256;
        int r = c >> 3, o = (c & 7) * 8;
        *(uint4*)(&As[r * LDT + o]) = s.a[i];
    }
#pragma unroll
    for (int i = 0; i < 8; ++i) {
        int c = tid + i * 256;
        int k = c >> 5, n = (c & 31) * 4;
        Bs[(n + 0) * LDT + k] = (bf16_t)s.b[i].x;
        Bs[(n + 1) * LDT + k] = (bf16_t)s.b[i].y;
        Bs[(n + 2) * LDT + k] = (bf16_t)s.b[i].z;
        Bs[(n + 3) * LDT + k] = (bf16_t)s.b[i].w;
    }
}

__device__ __forceinline__ void compute_stage(
    const bf16_t* __restrict__ As, const bf16_t* __restrict__ Bs,
    v8f acc[4][4], int wm, int wn, int lm, int kb)
{
#pragma unroll
    for (int j = 0; j < 2; ++j) {            // two 16x16x32 K-steps per stage
        const int ko = j * 32 + kb;
        v16bf b[4];
#pragma unroll
        for (int nt = 0; nt < 4; ++nt)
            b[nt] = frag16(&Bs[(wn + nt * 16 + lm) * LDT + ko]);
#pragma unroll
        for (int mt = 0; mt < 4; ++mt) {
            v16bf a = frag16(&As[(wm + mt * 16 + lm) * LDT + ko]);
#pragma unroll
            for (int nt = 0; nt < 4; ++nt)
                acc[mt][nt] = __builtin_amdgcn_wmma_f32_16x16x32_bf16(
                    false, a, false, b[nt], (short)0, acc[mt][nt], false, false);
        }
    }
}

// ---------------------------------------------------------------------------
// GEMM1: H1 = gelu( Xbf[4096,1024] @ W1e[1024,4096] + b1e )  (bf16 out)
// ---------------------------------------------------------------------------
__global__ __launch_bounds__(256) void moe_gemm1(
    const bf16_t* __restrict__ A, const float* __restrict__ W1,
    const float* __restrict__ b1, bf16_t* __restrict__ H1)
{
    __shared__ __align__(16) bf16_t As[2][BM * LDT];
    __shared__ __align__(16) bf16_t Bs[2][BN * LDT];

    const int tid = threadIdx.x;
    const int wv  = tid >> 5, ln = tid & 31;
    const int wm  = (wv & 3) * 64;
    const int wn  = (wv >> 2) * 64;
    const int row0 = blockIdx.x * BM;
    const int col0 = blockIdx.y * BN;
    const int lm  = ln & 15;
    const int kb  = (ln >> 4) * 8;

    v8f acc[4][4];
#pragma unroll
    for (int i = 0; i < 4; ++i)
#pragma unroll
        for (int j = 0; j < 4; ++j) {
            v8f z = {0.f,0.f,0.f,0.f,0.f,0.f,0.f,0.f};
            acc[i][j] = z;
        }

    StageRegs sr;
    load_stage(sr, A, HID, W1, INTER_, row0, col0, 0, tid);
    store_stage(sr, As[0], Bs[0], tid);
    __syncthreads();

    const int KT = HID / BK;                 // 16
    for (int kt = 0; kt < KT; ++kt) {
        const int cur = kt & 1;
        const bool has_next = (kt + 1 < KT);
        if (has_next)
            load_stage(sr, A, HID, W1, INTER_, row0, col0, (kt + 1) * BK, tid);
        compute_stage(As[cur], Bs[cur], acc, wm, wn, lm, kb);
        if (has_next)
            store_stage(sr, As[cur ^ 1], Bs[cur ^ 1], tid);
        __syncthreads();
    }

#pragma unroll
    for (int mt = 0; mt < 4; ++mt) {
        int rbase = row0 + wm + mt * 16 + ((ln >> 4) ? 8 : 0);
#pragma unroll
        for (int nt = 0; nt < 4; ++nt) {
            int col = col0 + wn + nt * 16 + lm;
            float bias = b1[col];
#pragma unroll
            for (int r = 0; r < 8; ++r) {
                float v = acc[mt][nt][r] + bias;
                H1[(size_t)(rbase + r) * INTER_ + col] = (bf16_t)gelu_f(v);
            }
        }
    }
}

// ---------------------------------------------------------------------------
// GEMM2: out (+)= combine[:,e] * ( H1[4096,4096] @ W2e[4096,1024] + b2e )
// ---------------------------------------------------------------------------
__global__ __launch_bounds__(256) void moe_gemm2(
    const bf16_t* __restrict__ A, const float* __restrict__ W2,
    const float* __restrict__ b2, const float* __restrict__ combine,
    const int expert, const int first, float* __restrict__ out)
{
    __shared__ __align__(16) bf16_t As[2][BM * LDT];
    __shared__ __align__(16) bf16_t Bs[2][BN * LDT];

    const int tid = threadIdx.x;
    const int wv  = tid >> 5, ln = tid & 31;
    const int wm  = (wv & 3) * 64;
    const int wn  = (wv >> 2) * 64;
    const int row0 = blockIdx.x * BM;
    const int col0 = blockIdx.y * BN;
    const int lm  = ln & 15;
    const int kb  = (ln >> 4) * 8;

    v8f acc[4][4];
#pragma unroll
    for (int i = 0; i < 4; ++i)
#pragma unroll
        for (int j = 0; j < 4; ++j) {
            v8f z = {0.f,0.f,0.f,0.f,0.f,0.f,0.f,0.f};
            acc[i][j] = z;
        }

    StageRegs sr;
    load_stage(sr, A, INTER_, W2, HID, row0, col0, 0, tid);
    store_stage(sr, As[0], Bs[0], tid);
    __syncthreads();

    const int KT = INTER_ / BK;              // 64
    for (int kt = 0; kt < KT; ++kt) {
        const int cur = kt & 1;
        const bool has_next = (kt + 1 < KT);
        if (has_next)
            load_stage(sr, A, INTER_, W2, HID, row0, col0, (kt + 1) * BK, tid);
        compute_stage(As[cur], Bs[cur], acc, wm, wn, lm, kb);
        if (has_next)
            store_stage(sr, As[cur ^ 1], Bs[cur ^ 1], tid);
        __syncthreads();
    }

#pragma unroll
    for (int mt = 0; mt < 4; ++mt) {
        int rbase = row0 + wm + mt * 16 + ((ln >> 4) ? 8 : 0);
#pragma unroll
        for (int nt = 0; nt < 4; ++nt) {
            int col = col0 + wn + nt * 16 + lm;
            float bias = b2[col];
#pragma unroll
            for (int r = 0; r < 8; ++r) {
                int row = rbase + r;
                float g = combine[row * NEXP + expert];
                float v = g * (acc[mt][nt][r] + bias);
                float* po = &out[(size_t)row * HID + col];
                *po = first ? v : (*po + v);
            }
        }
    }
}

// ---------------------------------------------------------------------------
extern "C" void kernel_launch(void* const* d_in, const int* in_sizes, int n_in,
                              void* d_out, int out_size, void* d_ws, size_t ws_size,
                              hipStream_t stream) {
    const float* x  = (const float*)d_in[0];
    const float* Wg = (const float*)d_in[1];
    const float* w1 = (const float*)d_in[2];
    const float* b1 = (const float*)d_in[3];
    const float* w2 = (const float*)d_in[4];
    const float* b2 = (const float*)d_in[5];
    float* out = (float*)d_out;

    char* ws = (char*)d_ws;
    bf16_t* Xbf     = (bf16_t*)ws;                                 // 8 MB
    bf16_t* H1      = (bf16_t*)(ws + (size_t)T_TOK * HID * 2);     // 32 MB
    float*  combine = (float*)(ws + (size_t)T_TOK * HID * 2
                                  + (size_t)T_TOK * INTER_ * 2);   // 128 KB
    float*  probs   = combine + (size_t)T_TOK * NEXP;              // 128 KB

    moe_gating<<<T_TOK, 256, 0, stream>>>(x, Wg, Xbf, combine, probs);
    moe_aux<<<1, 256, 0, stream>>>(probs, combine, out + (size_t)T_TOK * HID);

    for (int e = 0; e < NEXP; ++e) {
        moe_gemm1<<<dim3(T_TOK / BM, INTER_ / BN), 256, 0, stream>>>(
            Xbf, w1 + (size_t)e * HID * INTER_, b1 + (size_t)e * INTER_, H1);
        moe_gemm2<<<dim3(T_TOK / BM, HID / BN), 256, 0, stream>>>(
            H1, w2 + (size_t)e * INTER_ * HID, b2 + (size_t)e * HID,
            combine, e, (e == 0) ? 1 : 0, out);
    }
}
